// CodeAttention_40527311405160
// MI455X (gfx1250) — compile-verified, matrix operand order checked by reading, not run on hardware
//
#include <hip/hip_runtime.h>
#include <hip/hip_bf16.h>
#include <math.h>

typedef __bf16 bf16_t;
typedef __attribute__((ext_vector_type(16))) __bf16 v16bf;
typedef __attribute__((ext_vector_type(8)))  float  v8f;
typedef int i32x4 __attribute__((vector_size(16)));

#define B_ 4
#define S_ 2048
#define E_ 768
#define H_ 12
#define D_ 64

#define AS1 __attribute__((address_space(1)))
#define AS3 __attribute__((address_space(3)))

#if __has_builtin(__builtin_amdgcn_global_load_async_to_lds_b128)
#define HAS_ASYNC_LDS 1
#else
#define HAS_ASYNC_LDS 0
#endif

static __device__ inline void wait_async0() {
#if HAS_ASYNC_LDS
#if __has_builtin(__builtin_amdgcn_s_wait_asynccnt)
    __builtin_amdgcn_s_wait_asynccnt(0);
#else
    asm volatile("s_wait_asynccnt 0" ::: "memory");
#endif
#endif
}

static __device__ inline void async_copy16(const char* gsrc, char* ldst) {
#if HAS_ASYNC_LDS
    char* g = const_cast<char*>(gsrc);
    __builtin_amdgcn_global_load_async_to_lds_b128(
        (AS1 i32x4*)g, (AS3 i32x4*)ldst, 0, 0);
#else
    *(uint4*)ldst = *(const uint4*)gsrc;
#endif
}

static __device__ inline v8f wmma_bf16(v16bf a, v16bf b, v8f c) {
    return __builtin_amdgcn_wmma_f32_16x16x32_bf16(
        /*neg_a=*/false, a, /*neg_b=*/false, b,
        /*c_mod=*/(short)0, c, /*reuse_a=*/false, /*reuse_b=*/false);
}

// A/B fragment with contiguous K in memory (bf16 source).
// ISA 16-bit 16x32 layout: elements {0..7} -> K=koff+0..7, {8..15} -> K=koff+16..23,
// koff = 8 for lanes 16..31. Two contiguous 16-byte runs per lane.
static __device__ inline v16bf load_frag_bf16_contig(const bf16_t* p, int lane) {
    const int koff = (lane & 16) ? 8 : 0;
    v16bf r;
#pragma unroll
    for (int i = 0; i < 8; ++i) r[i] = p[koff + i];
#pragma unroll
    for (int i = 0; i < 8; ++i) r[8 + i] = p[koff + 16 + i];
    return r;
}

// ---------------- Layout kernels: fp32 -> bf16 (and transpose) ---------------
__global__ __launch_bounds__(256) void cvt_bf16_kernel(
    const float* __restrict__ in, bf16_t* __restrict__ out, long n) {
    const long i = ((long)blockIdx.x * 256 + threadIdx.x) * 4;
    if (i + 3 < n) {
        const float4 v = *(const float4*)(in + i);
        out[i]     = (bf16_t)v.x;
        out[i + 1] = (bf16_t)v.y;
        out[i + 2] = (bf16_t)v.z;
        out[i + 3] = (bf16_t)v.w;
    }
}

// in: [K, N] fp32 row-major  ->  out: [N, K] bf16 row-major (contiguous-K columns)
__global__ __launch_bounds__(256) void cvtT_bf16_kernel(
    const float* __restrict__ in, bf16_t* __restrict__ out, int K, int N) {
    const long t = (long)blockIdx.x * 256 + threadIdx.x;
    if (t < (long)N * K) {
        const int nn = (int)(t / K);
        const int kk = (int)(t % K);
        out[t] = (bf16_t)in[(long)kk * N + nn];
    }
}

// ---------------- Kernel 1: QKV GEMM + bias + head-split reshape -------------
__global__ __launch_bounds__(128) void qkv_gemm_kernel(
    const bf16_t* __restrict__ xb, const bf16_t* __restrict__ wt,
    const float* __restrict__ bqkv,
    bf16_t* __restrict__ qb, bf16_t* __restrict__ kb, bf16_t* __restrict__ vb) {
    const int lane  = threadIdx.x & 31;
    const int wid   = blockIdx.x * 4 + (threadIdx.x >> 5);
    const int NT    = (3 * E_) / 16;   // 144
    const int mTile = wid / NT;
    const int nTile = wid % NT;
    const int n     = nTile * 16 + (lane & 15);

    v8f c;
    const float bias = bqkv[n];
#pragma unroll
    for (int i = 0; i < 8; ++i) c[i] = bias;

    const long mBase = (long)mTile * 16;
    const bf16_t* ap = xb + (mBase + (lane & 15)) * E_;
    const bf16_t* bp = wt + (long)n * E_;
#pragma unroll 4
    for (int k0 = 0; k0 < E_; k0 += 32) {
        __builtin_prefetch(ap + k0 + 64, 0, 1);
        __builtin_prefetch(bp + k0 + 64, 0, 1);
        v16bf a = load_frag_bf16_contig(ap + k0, lane);
        v16bf b = load_frag_bf16_contig(bp + k0, lane);
        c = wmma_bf16(a, b, c);
    }

    const int which = n / E_;          // uniform across the 16-wide tile
    const int e = n - which * E_;
    const int h = e >> 6;
    const int d = e & 63;
#pragma unroll
    for (int r = 0; r < 8; ++r) {
        const long m  = mBase + r + ((lane >> 4) << 3);
        const int  bb = (int)(m >> 11);
        const int  s  = (int)(m & 2047);
        const bf16_t val = (bf16_t)c[r];
        if (which == 0)      qb[(((long)bb * H_ + h) * S_ + s) * D_ + d] = val;
        else if (which == 1) kb[(((long)bb * H_ + h) * S_ + s) * D_ + d] = val;
        else                 vb[(((long)bb * H_ + h) * D_ + d) * S_ + s] = val;
    }
}

// ---------------- Kernel 2: flash attention with async K/V LDS staging -------
// Block = 4 waves sharing one (b,h); each wave owns one 16-query tile.
// Per 32-key step: stage K tile [32 x 64] and V tile [64 x 32] (bf16) into LDS
// via GLOBAL_LOAD_ASYNC_TO_LDS_B128 (double buffered), then consume from LDS.
__global__ __launch_bounds__(128) void attn_kernel(
    const bf16_t* __restrict__ qb, const bf16_t* __restrict__ kb,
    const bf16_t* __restrict__ vb, const int* __restrict__ mask,
    bf16_t* __restrict__ attnout) {
    __shared__ bf16_t ldsK[2][32 * 64];    // [key][d]   4KB per buffer
    __shared__ bf16_t ldsV[2][64 * 32];    // [d][key]   4KB per buffer
    __shared__ bf16_t pbuf[4][16 * 32];    // wave-private P staging

    const int tid   = threadIdx.x;
    const int lane  = tid & 31;
    const int wslot = tid >> 5;
    const int bh    = blockIdx.x >> 5;         // 48 (b,h) pairs
    const int qgrp  = blockIdx.x & 31;         // 32 groups of 4 query tiles
    const int qt    = qgrp * 4 + wslot;
    const int b     = bh / H_;
    const int h     = bh % H_;

    const bf16_t* Q = qb + (long)bh * S_ * D_;
    const bf16_t* K = kb + (long)bh * S_ * D_;
    const bf16_t* V = vb + (long)bh * D_ * S_;   // [D, S]
    const int* mrow = mask + (long)b * S_;

    const int qrow = qt * 16 + (lane & 15);
    const v16bf qa0 = load_frag_bf16_contig(Q + (long)qrow * D_, lane);
    const v16bf qa1 = load_frag_bf16_contig(Q + (long)qrow * D_ + 32, lane);

    v8f o[4] = {};
    float mrun[8], lrun[8];
#pragma unroll
    for (int r = 0; r < 8; ++r) { mrun[r] = -3.0e38f; lrun[r] = 0.0f; }

    bf16_t* pl = pbuf[wslot];
    const int arow  = lane & 15;
    const int chalf = (lane >> 4) << 3;

    // ---- cooperative staging of one 32-key tile into LDS buffer `buf` ----
    auto stage = [&](int kt, int buf) {
        // K tile: 32 consecutive rows of 64 bf16 -> contiguous 4KB in global.
        const char* gk  = (const char*)(K + (long)kt * D_);
        char*       lk  = (char*)ldsK[buf];
        const char* gvb = (const char*)V;
        char*       lv  = (char*)ldsV[buf];
#pragma unroll
        for (int c = tid; c < 256; c += 128) {
            async_copy16(gk + c * 16, lk + c * 16);
        }
#pragma unroll
        for (int c = tid; c < 256; c += 128) {
            const int row = c >> 2, part = c & 3;   // d row, 16B part of 64B row
            async_copy16(gvb + ((long)row * S_ + kt) * 2 + part * 16, lv + c * 16);
        }
    };

    stage(0, 0);   // prologue

    for (int it = 0; it < S_ / 32; ++it) {
        const int kt  = it * 32;
        const int cur = it & 1;

        wait_async0();        // my async loads for tile `it` have landed
        __syncthreads();      // everyone's loads done + buffer cur^1 free
        if (kt + 32 < S_) stage(kt + 32, cur ^ 1);

        const bf16_t* Kl = ldsK[cur];
        const bf16_t* Vl = ldsV[cur];

        // ---- scores: S = Q(16x64) @ K^T(64x32), two 16x16 N-subtiles ----
        const int kl0 = lane & 15;          // local key column, subtile 0
        v16bf kb0a = load_frag_bf16_contig(Kl + kl0 * 64, lane);
        v16bf kb0b = load_frag_bf16_contig(Kl + kl0 * 64 + 32, lane);
        v16bf kb1a = load_frag_bf16_contig(Kl + (kl0 + 16) * 64, lane);
        v16bf kb1b = load_frag_bf16_contig(Kl + (kl0 + 16) * 64 + 32, lane);
        v8f s0 = {}, s1 = {};
        s0 = wmma_bf16(qa0, kb0a, s0);
        s0 = wmma_bf16(qa1, kb0b, s0);
        s1 = wmma_bf16(qa0, kb1a, s1);
        s1 = wmma_bf16(qa1, kb1b, s1);

        // scale 1/sqrt(64) + additive padding mask (per key column)
        const bool m0 = mrow[kt + kl0] != 0;
        const bool m1 = mrow[kt + kl0 + 16] != 0;
#pragma unroll
        for (int r = 0; r < 8; ++r) {
            s0[r] = m0 ? s0[r] * 0.125f : -1.0e9f;
            s1[r] = m1 ? s1[r] * 0.125f : -1.0e9f;
        }

        // ---- online softmax: row stats live per C-fragment element ----
        float tm[8], ts[8];
#pragma unroll
        for (int r = 0; r < 8; ++r) tm[r] = fmaxf(s0[r], s1[r]);
        for (int off = 1; off < 16; off <<= 1) {
#pragma unroll
            for (int r = 0; r < 8; ++r)
                tm[r] = fmaxf(tm[r], __shfl_xor(tm[r], off, 32));
        }
#pragma unroll
        for (int r = 0; r < 8; ++r) {
            const float mnew = fmaxf(mrun[r], tm[r]);
            const float corr = __expf(mrun[r] - mnew);
            mrun[r] = mnew;
            const float p0 = __expf(s0[r] - mnew);
            const float p1 = __expf(s1[r] - mnew);
            s0[r] = p0; s1[r] = p1;
            ts[r] = p0 + p1;
            o[0][r] *= corr; o[1][r] *= corr; o[2][r] *= corr; o[3][r] *= corr;
            lrun[r] *= corr;
        }
        for (int off = 1; off < 16; off <<= 1) {
#pragma unroll
            for (int r = 0; r < 8; ++r) ts[r] += __shfl_xor(ts[r], off, 32);
        }
#pragma unroll
        for (int r = 0; r < 8; ++r) lrun[r] += ts[r];

        // ---- stage P (C-layout) into LDS row-major 16x32 bf16 ----
#pragma unroll
        for (int r = 0; r < 8; ++r) {
            pl[(r + chalf) * 32 + kl0]      = (bf16_t)s0[r];
            pl[(r + chalf) * 32 + kl0 + 16] = (bf16_t)s1[r];
        }
        v16bf pa = load_frag_bf16_contig(pl + arow * 32, lane);

        // ---- O += P(16x32) @ V(32x64), four 16x16 N-subtiles ----
#pragma unroll
        for (int j = 0; j < 4; ++j) {
            v16bf vf = load_frag_bf16_contig(Vl + (j * 16 + kl0) * 32, lane);
            o[j] = wmma_bf16(pa, vf, o[j]);
        }
    }

    // ---- normalize and write attn output as [B, S, E] bf16 ----
    float inv[8];
#pragma unroll
    for (int r = 0; r < 8; ++r) inv[r] = 1.0f / fmaxf(lrun[r], 1.0e-20f);
#pragma unroll
    for (int j = 0; j < 4; ++j) {
        const int eo = h * 64 + j * 16 + (lane & 15);
#pragma unroll
        for (int r = 0; r < 8; ++r) {
            const int srow = qt * 16 + r + chalf;
            attnout[((long)b * S_ + srow) * E_ + eo] = (bf16_t)(o[j][r] * inv[r]);
        }
    }
}

// ---------------- Kernel 3: output projection GEMM + bias --------------------
__global__ __launch_bounds__(128) void proj_gemm_kernel(
    const bf16_t* __restrict__ ain, const bf16_t* __restrict__ wt,
    const float* __restrict__ bp, float* __restrict__ out) {
    const int lane  = threadIdx.x & 31;
    const int wid   = blockIdx.x * 4 + (threadIdx.x >> 5);
    const int NT    = E_ / 16;        // 48
    const int mTile = wid / NT;
    const int nTile = wid % NT;
    const int n     = nTile * 16 + (lane & 15);

    v8f c;
    const float bias = bp[n];
#pragma unroll
    for (int i = 0; i < 8; ++i) c[i] = bias;

    const long mBase = (long)mTile * 16;
    const bf16_t* ap  = ain + (mBase + (lane & 15)) * E_;
    const bf16_t* bpw = wt + (long)n * E_;
#pragma unroll 4
    for (int k0 = 0; k0 < E_; k0 += 32) {
        __builtin_prefetch(ap + k0 + 64, 0, 1);
        __builtin_prefetch(bpw + k0 + 64, 0, 1);
        v16bf a = load_frag_bf16_contig(ap + k0, lane);
        v16bf b = load_frag_bf16_contig(bpw + k0, lane);
        c = wmma_bf16(a, b, c);
    }
#pragma unroll
    for (int r = 0; r < 8; ++r) {
        const long m = mBase + r + ((lane >> 4) << 3);
        out[m * E_ + n] = c[r];
    }
}

extern "C" void kernel_launch(void* const* d_in, const int* in_sizes, int n_in,
                              void* d_out, int out_size, void* d_ws, size_t ws_size,
                              hipStream_t stream) {
    const float* x     = (const float*)d_in[0];
    const int*   maskp = (const int*)d_in[1];
    const float* Wqkv  = (const float*)d_in[2];
    const float* bqkv  = (const float*)d_in[3];
    const float* Wproj = (const float*)d_in[4];
    const float* bproj = (const float*)d_in[5];
    float* out = (float*)d_out;

    char* ws = (char*)d_ws;
    const size_t qkv_elems = (size_t)B_ * H_ * S_ * D_;   // 6,291,456
    const size_t x_elems   = (size_t)B_ * S_ * E_;        // 6,291,456
    size_t off = 0;
    bf16_t* qb      = (bf16_t*)(ws + off); off += qkv_elems * 2;
    bf16_t* kbuf    = (bf16_t*)(ws + off); off += qkv_elems * 2;
    bf16_t* vbuf    = (bf16_t*)(ws + off); off += qkv_elems * 2;
    bf16_t* xb      = (bf16_t*)(ws + off); off += x_elems * 2;
    bf16_t* attnout = (bf16_t*)(ws + off); off += x_elems * 2;
    bf16_t* wqkvT   = (bf16_t*)(ws + off); off += (size_t)(3 * E_) * E_ * 2;
    bf16_t* wprojT  = (bf16_t*)(ws + off); off += (size_t)E_ * E_ * 2;

    // ---- one-time layout/precision prep (bf16, weights transposed) ----
    cvt_bf16_kernel<<<(int)(x_elems / 1024), 256, 0, stream>>>(x, xb, (long)x_elems);
    cvtT_bf16_kernel<<<(3 * E_ * E_) / 256, 256, 0, stream>>>(Wqkv, wqkvT, E_, 3 * E_);
    cvtT_bf16_kernel<<<(E_ * E_) / 256, 256, 0, stream>>>(Wproj, wprojT, E_, E_);

    {   // 512 M-tiles x 144 N-tiles, 4 waves/block
        const int waves = (B_ * S_ / 16) * (3 * E_ / 16);
        qkv_gemm_kernel<<<waves / 4, 128, 0, stream>>>(xb, wqkvT, bqkv, qb, kbuf, vbuf);
    }
    {   // 48 (b,h) x 32 query-tile groups (4 waves each)
        attn_kernel<<<48 * 32, 128, 0, stream>>>(qb, kbuf, vbuf, maskp, attnout);
    }
    {   // 512 M-tiles x 48 N-tiles
        const int waves = (B_ * S_ / 16) * (E_ / 16);
        proj_gemm_kernel<<<waves / 4, 128, 0, stream>>>(attnout, wprojT, bproj, out);
    }
}
